// GatedDeltaNet_77902116815051
// MI455X (gfx1250) — matrix operand order repaired and checked by reading, not compiled, not run on hardware
//
#include <hip/hip_runtime.h>

typedef __bf16 bf16;
typedef __attribute__((ext_vector_type(16))) __bf16 v16bf;
typedef __attribute__((ext_vector_type(8)))  __bf16 v8bf;
typedef __attribute__((ext_vector_type(4)))  __bf16 v4bf;
typedef __attribute__((ext_vector_type(8)))  float  v8f;

// ---------------------------------------------------------------------------
// WMMA helpers (CDNA5 gfx1250, wave32). Layouts per CDNA5 ISA.
// ---------------------------------------------------------------------------
static __device__ __forceinline__ v8f wmma_bf(v16bf a, v16bf b, v8f c) {
  return __builtin_amdgcn_wmma_f32_16x16x32_bf16(false, a, false, b, (short)0, c,
                                                 false, false);
}

// A-fragment (16x32, M x K) from a row-major bf16 row pointer.
// lane half g: element e -> K = k0+16g+e  (two b128 loads)
static __device__ __forceinline__ v16bf afrag_row(const bf16* row, int k0, int g) {
  v8bf lo = *(const v8bf*)(row + k0 + 16 * g);
  v8bf hi = *(const v8bf*)(row + k0 + 16 * g + 8);
  v16bf r;
#pragma unroll
  for (int i = 0; i < 8; ++i) { r[i] = lo[i]; r[8 + i] = hi[i]; }
  return r;
}

// B-fragment (32x16, K x N) where memory holds B^T row-major:
// row = &BT[n][0]; element e -> B[k0+16g+e][n] = BT[n][k0+16g+e]
static __device__ __forceinline__ v16bf bfrag_rowT(const bf16* row, int k0, int g) {
  v8bf lo = *(const v8bf*)(row + k0 + 16 * g);
  v8bf hi = *(const v8bf*)(row + k0 + 16 * g + 8);
  v16bf r;
#pragma unroll
  for (int i = 0; i < 8; ++i) { r[i] = lo[i]; r[8 + i] = hi[i]; }
  return r;
}

// fragments from 32x32 f32 LDS matrices (converted to bf16 on the fly)
static __device__ __forceinline__ v16bf afrag32_f32(const float* M, int m, int g) {
  v16bf r;
#pragma unroll
  for (int e = 0; e < 16; ++e) {
    int k = 16 * g + e;
    r[e] = (bf16)M[m * 32 + k];
  }
  return r;
}
static __device__ __forceinline__ v16bf bfrag32_f32(const float* M, int n, int g) {
  v16bf r;
#pragma unroll
  for (int e = 0; e < 16; ++e) r[e] = (bf16)M[(16 * g + e) * 32 + n];
  return r;
}

// one 16x16 tile of a 32x32x32 f32-in/out GEMM, executed by wave w4 in [0,4)
static __device__ __forceinline__ void gemm32(const float* A, const float* B,
                                              const float* Cin, float* D, int w4,
                                              int lane) {
  const int g = lane >> 4, ln = lane & 15;
  const int mt = (w4 >> 1) * 16, nt = (w4 & 1) * 16;
  v8f acc = {};
  if (Cin) {
#pragma unroll
    for (int r = 0; r < 8; ++r) acc[r] = Cin[(mt + 8 * g + r) * 32 + nt + ln];
  }
  v16bf a = afrag32_f32(A, mt + ln, g);
  v16bf b = bfrag32_f32(B, nt + ln, g);
  acc = wmma_bf(a, b, acc);
#pragma unroll
  for (int r = 0; r < 8; ++r) D[(mt + 8 * g + r) * 32 + nt + ln] = acc[r];
}

// ---------------------------------------------------------------------------
// f32 -> bf16 conversion (vectorized x4)
// ---------------------------------------------------------------------------
__global__ __launch_bounds__(256) void cvt_kernel(const float* __restrict__ in,
                                                  bf16* __restrict__ out, int n4) {
  int i = blockIdx.x * 256 + threadIdx.x;
  if (i < n4) {
    float4 v = ((const float4*)in)[i];
    v4bf o;
    o[0] = (bf16)v.x; o[1] = (bf16)v.y; o[2] = (bf16)v.z; o[3] = (bf16)v.w;
    ((v4bf*)out)[i] = o;
  }
}

// ---------------------------------------------------------------------------
// Tiled GEMM: C[M,N] f32 = A[M,K] bf16 @ W[N,K]^T bf16   (W row-major == B^T)
// 256 threads = 8 waves. Block tile 128x128, k-step 32, double-buffered LDS.
// Wave w owns M-strip [16w,16w+16): 1 A-fragment vs 8 B-fragments -> 8 WMMA
// per 18 ds_load_b128 (~64 flops/byte of L2 traffic).
// Requires M%128==0, N%128==0, K%32==0.
// ---------------------------------------------------------------------------
__global__ __launch_bounds__(256) void gemm_bf16(const bf16* __restrict__ A,
                                                 const bf16* __restrict__ W,
                                                 float* __restrict__ C, int M, int N,
                                                 int K) {
  __shared__ __attribute__((aligned(16))) bf16 sA[2][128 * 32];
  __shared__ __attribute__((aligned(16))) bf16 sB[2][128 * 32];
  const int tid = threadIdx.x;
  const int wv = tid >> 5, lane = tid & 31;
  const int g = lane >> 4, ln = lane & 15;
  const int m0 = blockIdx.y * 128;
  const int n0 = blockIdx.x * 128;
  const int nk = K >> 5;

  // staging: 512 chunks of 8 bf16 (16B); thread handles chunks tid and tid+256
  const int r0 = tid >> 2, q0 = tid & 3;
  const bf16* gA0 = A + (size_t)(m0 + r0) * K + 8 * q0;
  const bf16* gA1 = A + (size_t)(m0 + r0 + 64) * K + 8 * q0;
  const bf16* gB0 = W + (size_t)(n0 + r0) * K + 8 * q0;
  const bf16* gB1 = W + (size_t)(n0 + r0 + 64) * K + 8 * q0;
  const int ls0 = r0 * 32 + 8 * q0;
  const int ls1 = (r0 + 64) * 32 + 8 * q0;

  v8f zero = {};
  v8f acc[8];
#pragma unroll
  for (int i = 0; i < 8; ++i) acc[i] = zero;

  v8bf ra0 = *(const v8bf*)(gA0);
  v8bf ra1 = *(const v8bf*)(gA1);
  v8bf rb0 = *(const v8bf*)(gB0);
  v8bf rb1 = *(const v8bf*)(gB1);
  *(v8bf*)(sA[0] + ls0) = ra0;
  *(v8bf*)(sA[0] + ls1) = ra1;
  *(v8bf*)(sB[0] + ls0) = rb0;
  *(v8bf*)(sB[0] + ls1) = rb1;
  __syncthreads();

  for (int ks = 0; ks < nk; ++ks) {
    const int cur = ks & 1;
    const bool more = (ks + 1) < nk;
    if (more) {  // prefetch next panel into registers while computing
      size_t off = (size_t)(ks + 1) * 32;
      ra0 = *(const v8bf*)(gA0 + off);
      ra1 = *(const v8bf*)(gA1 + off);
      rb0 = *(const v8bf*)(gB0 + off);
      rb1 = *(const v8bf*)(gB1 + off);
    }
    v16bf afr = afrag_row(sA[cur] + (16 * wv + ln) * 32, 0, g);
#pragma unroll
    for (int nt = 0; nt < 8; ++nt) {
      v16bf bfr = bfrag_rowT(sB[cur] + (nt * 16 + ln) * 32, 0, g);
      acc[nt] = wmma_bf(afr, bfr, acc[nt]);
    }
    if (more) {
      const int nxt = cur ^ 1;
      *(v8bf*)(sA[nxt] + ls0) = ra0;
      *(v8bf*)(sA[nxt] + ls1) = ra1;
      *(v8bf*)(sB[nxt] + ls0) = rb0;
      *(v8bf*)(sB[nxt] + ls1) = rb1;
    }
    __syncthreads();
  }

#pragma unroll
  for (int nt = 0; nt < 8; ++nt) {
    float* crow = C + (size_t)(m0 + 16 * wv + 8 * g) * N + n0 + nt * 16 + ln;
#pragma unroll
    for (int r = 0; r < 8; ++r) crow[(size_t)r * N] = acc[nt][r];
  }
}

// ---------------------------------------------------------------------------
// gates: dot(x, gk_w[h]) -> log-alpha (logsigmoid/16) ; dot(x, b_w[h]) -> beta
// ---------------------------------------------------------------------------
__global__ __launch_bounds__(256) void gates_kernel(const float* __restrict__ x,
                                                    const float* __restrict__ gk_w,
                                                    const float* __restrict__ b_w,
                                                    float* __restrict__ ga,
                                                    float* __restrict__ be) {
  int wid = (blockIdx.x * 256 + threadIdx.x) >> 5;
  int lane = threadIdx.x & 31;
  if (wid >= 4096) return;
  const float* xr = x + (size_t)wid * 2048;
  float accA[16], accB[16];
#pragma unroll
  for (int h = 0; h < 16; ++h) { accA[h] = 0.f; accB[h] = 0.f; }
  for (int k = lane; k < 2048; k += 32) {
    float xv = xr[k];
#pragma unroll
    for (int h = 0; h < 16; ++h) {
      accA[h] += xv * gk_w[h * 2048 + k];
      accB[h] += xv * b_w[h * 2048 + k];
    }
  }
#pragma unroll
  for (int h = 0; h < 16; ++h) {
    float a = accA[h], b = accB[h];
#pragma unroll
    for (int off = 16; off > 0; off >>= 1) {
      a += __shfl_xor(a, off, 32);
      b += __shfl_xor(b, off, 32);
    }
    if (lane == 0) {
      float ls = (fminf(a, 0.f) - log1pf(expf(-fabsf(a)))) * (1.f / 16.f);
      ga[(size_t)wid * 16 + h] = ls;
      be[(size_t)wid * 16 + h] = 1.f / (1.f + expf(-b));
    }
  }
}

// ---------------------------------------------------------------------------
// causal depthwise conv (K=4) + SiLU over (B,S,3072)
// ---------------------------------------------------------------------------
__global__ __launch_bounds__(256) void conv_silu(const float* __restrict__ in,
                                                 const float* __restrict__ w,
                                                 float* __restrict__ out) {
  int idx = blockIdx.x * 256 + threadIdx.x;  // < 2*2048*3072
  int c = idx % 3072;
  int s = (idx / 3072) & 2047;
  float acc = 0.f;
#pragma unroll
  for (int i = 0; i < 4; ++i) {
    int sp = s - 3 + i;
    if (sp >= 0) acc += w[c * 4 + i] * in[idx + (i - 3) * 3072];
  }
  out[idx] = acc / (1.f + expf(-acc));
}

// ---------------------------------------------------------------------------
// split qkv, L2-normalize q (x DK^-0.5) and k, copy v. one wave per (row,slot)
// ---------------------------------------------------------------------------
__global__ __launch_bounds__(256) void prep_kernel(const float* __restrict__ qkvc,
                                                   float* __restrict__ qb,
                                                   float* __restrict__ kb,
                                                   float* __restrict__ vb) {
  int wid = (blockIdx.x * 256 + threadIdx.x) >> 5;
  int lane = threadIdx.x & 31;
  int slot = wid % 24;
  int row = wid / 24;
  if (row >= 4096) return;
  const float* src;
  float* dst;
  bool do_norm = true;
  float scale = 1.f;
  if (slot < 16) {
    src = qkvc + (size_t)row * 3072 + slot * 128;
    dst = qb + ((size_t)row * 16 + slot) * 128;
    scale = 0.08838834764831845f;  // 128^-0.5
  } else if (slot < 20) {
    int hh = slot - 16;
    src = qkvc + (size_t)row * 3072 + 2048 + hh * 128;
    dst = kb + ((size_t)row * 4 + hh) * 128;
  } else {
    int hh = slot - 20;
    src = qkvc + (size_t)row * 3072 + 2560 + hh * 128;
    dst = vb + ((size_t)row * 4 + hh) * 128;
    do_norm = false;
  }
  float vals[4];
  float ss = 0.f;
#pragma unroll
  for (int i = 0; i < 4; ++i) {
    vals[i] = src[lane + 32 * i];
    ss += vals[i] * vals[i];
  }
  if (do_norm) {
#pragma unroll
    for (int off = 16; off > 0; off >>= 1) ss += __shfl_xor(ss, off, 32);
    float rs = rsqrtf(ss + 1e-6f) * scale;
#pragma unroll
    for (int i = 0; i < 4; ++i) vals[i] *= rs;
  }
#pragma unroll
  for (int i = 0; i < 4; ++i) dst[lane + 32 * i] = vals[i];
}

// ---------------------------------------------------------------------------
// Chunked gated delta rule. One block per (b,h); chunk C=32; state 128x128 in
// registers laid out as WMMA B-fragments (wave w owns cols [16w,16w+16)).
// ---------------------------------------------------------------------------
__global__ __launch_bounds__(256) void scan_kernel(const float* __restrict__ qb,
                                                   const float* __restrict__ kb,
                                                   const float* __restrict__ vb,
                                                   const float* __restrict__ ga,
                                                   const float* __restrict__ be,
                                                   float* __restrict__ ob) {
  const int bh = blockIdx.x;
  const int b = bh >> 4, h = bh & 15, hk = h >> 2;
  const int tid = threadIdx.x;
  const int wv = tid >> 5, lane = tid & 31;
  const int g = lane >> 4, ln = lane & 15;

  __shared__ __attribute__((aligned(16))) bf16 sK[32 * 128];   // (t, dk)
  __shared__ __attribute__((aligned(16))) bf16 sKT[128 * 32];  // (dk, t)
  __shared__ __attribute__((aligned(16))) bf16 sQ[32 * 128];   // (t, dk)
  __shared__ __attribute__((aligned(16))) bf16 sBT[128 * 32];  // B^T (dv, t)
  __shared__ __attribute__((aligned(16))) bf16 sDT[128 * 32];  // Delta^T (dv, t)
  __shared__ float sM[32 * 32];                                // N (strict lower)
  __shared__ float sP[32 * 32];                                // masked QK^T decay
  __shared__ float sT0[32 * 32], sT1[32 * 32], sW[32 * 32];
  __shared__ float sG[32], sBe[32], sEg[32], sEd[32];

  float st[4][16];  // state B-fragments: st[i][e] = S[32i+16g+e][16wv+ln]
#pragma unroll
  for (int i = 0; i < 4; ++i)
#pragma unroll
    for (int e = 0; e < 16; ++e) st[i][e] = 0.f;

#pragma unroll 1
  for (int ch = 0; ch < 64; ++ch) {
    const int t0 = ch * 32;
    // ---- stage chunk into LDS (bf16) ----
#pragma unroll
    for (int i = 0; i < 16; ++i) {
      int idx = tid + 256 * i;  // 0..4095
      int t = idx >> 7, d = idx & 127;
      size_t gr = (size_t)(b * 2048 + t0 + t);
      bf16 kv = (bf16)kb[gr * 512 + hk * 128 + d];
      sK[t * 128 + d] = kv;
      sKT[d * 32 + t] = kv;
      sQ[t * 128 + d] = (bf16)qb[gr * 2048 + h * 128 + d];
    }
    if (tid < 32) {
      size_t gr = (size_t)(b * 2048 + t0 + tid);
      sG[tid] = ga[gr * 16 + h];
      sBe[tid] = be[gr * 16 + h];
    }
    __syncthreads();
    if (tid == 0) {  // inclusive prefix of log-decay
      float acc = 0.f;
      for (int t = 0; t < 32; ++t) { acc += sG[t]; sG[t] = acc; }
    }
    __syncthreads();
    if (tid < 32) {
      sEg[tid] = __expf(sG[tid]);
      sEd[tid] = __expf(sG[31] - sG[tid]);
    }
    // ---- KK^T (waves 0-3 -> sM raw), QK^T (waves 4-7 -> sP raw) ----
    {
      int w4 = wv & 3;
      const bf16* Abase = (wv < 4) ? sK : sQ;
      float* Dst = (wv < 4) ? sM : sP;
      const int mt = (w4 >> 1) * 16, nt = (w4 & 1) * 16;
      v8f acc = {};
#pragma unroll
      for (int ks = 0; ks < 4; ++ks) {
        v16bf a = afrag_row(Abase + (mt + ln) * 128, ks * 32, g);
        v16bf bb = bfrag_rowT(sK + (nt + ln) * 128, ks * 32, g);  // B = K^T
        acc = wmma_bf(a, bb, acc);
      }
#pragma unroll
      for (int r = 0; r < 8; ++r) Dst[(mt + 8 * g + r) * 32 + nt + ln] = acc[r];
    }
    __syncthreads();
    // ---- decay-scale + mask; T0 = I - N ----
#pragma unroll
    for (int ii = 0; ii < 4; ++ii) {
      int idx = tid + 256 * ii;
      int i = idx >> 5, j = idx & 31;
      float m = 0.f, p = 0.f;
      if (j <= i) {
        float eij = __expf(sG[i] - sG[j]);
        p = eij * sP[idx];
        if (j < i) m = sBe[i] * eij * sM[idx];
      }
      sM[idx] = m;
      sP[idx] = p;
      sT0[idx] = ((i == j) ? 1.f : 0.f) - m;
    }
    __syncthreads();
    // ---- B = beta * (V - e^G * (K @ S0)), stored transposed bf16 ----
#pragma unroll
    for (int mt = 0; mt < 2; ++mt) {
      v8f acc = {};
#pragma unroll
      for (int ks = 0; ks < 4; ++ks) {
        v16bf a = afrag_row(sK + (mt * 16 + ln) * 128, ks * 32, g);
        v16bf bb;
#pragma unroll
        for (int e = 0; e < 16; ++e) bb[e] = (bf16)st[ks][e];
        acc = wmma_bf(a, bb, acc);
      }
      v8bf pack;
#pragma unroll
      for (int r = 0; r < 8; ++r) {
        int t = mt * 16 + 8 * g + r;
        float vvv = vb[(size_t)(b * 2048 + t0 + t) * 512 + hk * 128 + 16 * wv + ln];
        pack[r] = (bf16)(sBe[t] * (vvv - sEg[t] * acc[r]));
      }
      *(v8bf*)(sBT + (16 * wv + ln) * 32 + mt * 16 + 8 * g) = pack;
    }
    __syncthreads();
    // ---- T = (I-N)(I+N^2)(I+N^4)(I+N^8)(I+N^16) via WMMA ----
    if (wv < 4) gemm32(sM, sM, nullptr, sW, wv, lane);  // W = N^2
    __syncthreads();
    {
      float *Tc = sT0, *Tn = sT1, *Wc = sW, *Wn = sM;  // sM reusable now
#pragma unroll 1
      for (int it = 0; it < 4; ++it) {
        if (wv < 4) gemm32(Tc, Wc, Tc, Tn, wv, lane);                // T <- T*W + T
        else if (it < 3) gemm32(Wc, Wc, nullptr, Wn, wv - 4, lane);  // W <- W^2
        __syncthreads();
        float* t = Tc; Tc = Tn; Tn = t;
        t = Wc; Wc = Wn; Wn = t;
      }
      // ---- Delta = T @ B  -> sDT (transposed bf16) ----
#pragma unroll
      for (int mt = 0; mt < 2; ++mt) {
        v8f acc = {};
        v16bf a = afrag32_f32(Tc, mt * 16 + ln, g);
        v16bf bb = bfrag_rowT(sBT + (16 * wv + ln) * 32, 0, g);
        acc = wmma_bf(a, bb, acc);
        v8bf pack;
#pragma unroll
        for (int r = 0; r < 8; ++r) pack[r] = (bf16)acc[r];
        *(v8bf*)(sDT + (16 * wv + ln) * 32 + mt * 16 + 8 * g) = pack;
      }
    }
    __syncthreads();
    // ---- O = e^G ⊙ (Q @ S0) + (P∘mask) @ Delta ----
#pragma unroll
    for (int mt = 0; mt < 2; ++mt) {
      v8f acc = {};
#pragma unroll
      for (int ks = 0; ks < 4; ++ks) {
        v16bf a = afrag_row(sQ + (mt * 16 + ln) * 128, ks * 32, g);
        v16bf bb;
#pragma unroll
        for (int e = 0; e < 16; ++e) bb[e] = (bf16)st[ks][e];
        acc = wmma_bf(a, bb, acc);
      }
#pragma unroll
      for (int r = 0; r < 8; ++r) acc[r] *= sEg[mt * 16 + 8 * g + r];
      v16bf a2 = afrag32_f32(sP, mt * 16 + ln, g);
      v16bf b2 = bfrag_rowT(sDT + (16 * wv + ln) * 32, 0, g);
      acc = wmma_bf(a2, b2, acc);
#pragma unroll
      for (int r = 0; r < 8; ++r) {
        int t = mt * 16 + 8 * g + r;
        ob[(size_t)(b * 2048 + t0 + t) * 2048 + h * 128 + 16 * wv + ln] = acc[r];
      }
    }
    // ---- S' = e^{Gend} S0 + K̂^T @ Delta ; fold back into B-fragments ----
    {
      float eGend = sEg[31];
#pragma unroll
      for (int i = 0; i < 4; ++i) {
        v8f dd[2];
#pragma unroll
        for (int half = 0; half < 2; ++half) {
          int mrow = (2 * i + half) * 16 + ln;  // k-row of state
          const bf16* kt = sKT + mrow * 32;
          v8bf lo = *(const v8bf*)(kt + 16 * g);
          v8bf hi = *(const v8bf*)(kt + 16 * g + 8);
          v16bf a;
#pragma unroll
          for (int e = 0; e < 8; ++e) {
            a[e] = (bf16)((float)lo[e] * sEd[16 * g + e]);
            a[8 + e] = (bf16)((float)hi[e] * sEd[16 * g + 8 + e]);
          }
          v16bf bb = bfrag_rowT(sDT + (16 * wv + ln) * 32, 0, g);
          v8f z = {};
          dd[half] = wmma_bf(a, bb, z);
        }
        // C-layout (two stacked 16x16 tiles) -> B-fragment via lane xor 16
        float s0[8], s1[8];
#pragma unroll
        for (int r = 0; r < 8; ++r) {
          s0[r] = __shfl_xor(dd[0][r], 16, 32);
          s1[r] = __shfl_xor(dd[1][r], 16, 32);
        }
#pragma unroll
        for (int e = 0; e < 8; ++e) {
          float add_lo = g ? s1[e] : dd[0][e];
          float add_hi = g ? dd[1][e] : s0[e];
          st[i][e] = st[i][e] * eGend + add_lo;
          st[i][e + 8] = st[i][e + 8] * eGend + add_hi;
        }
      }
    }
    __syncthreads();  // protect LDS before next chunk's staging
  }
}

// ---------------------------------------------------------------------------
// per-head RMSNorm * out_norm_w * silu(g) -> bf16 rows for final GEMM
// ---------------------------------------------------------------------------
__global__ __launch_bounds__(256) void rms_gate_kernel(const float* __restrict__ o,
                                                       const float* __restrict__ gb,
                                                       const float* __restrict__ w,
                                                       bf16* __restrict__ y) {
  int wid = (blockIdx.x * 256 + threadIdx.x) >> 5;  // row*16 + h
  int lane = threadIdx.x & 31;
  if (wid >= 65536) return;
  int h = wid & 15, row = wid >> 4;
  const float* op = o + (size_t)row * 2048 + h * 128;
  const float* gp = gb + (size_t)row * 2048 + h * 128;
  float v[4], ss = 0.f;
#pragma unroll
  for (int i = 0; i < 4; ++i) {
    v[i] = op[lane + 32 * i];
    ss += v[i] * v[i];
  }
#pragma unroll
  for (int off = 16; off > 0; off >>= 1) ss += __shfl_xor(ss, off, 32);
  float rs = rsqrtf(ss * (1.f / 128.f) + 1e-5f);
#pragma unroll
  for (int i = 0; i < 4; ++i) {
    float gg = gp[lane + 32 * i];
    float sg = gg / (1.f + expf(-gg));
    y[(size_t)row * 2048 + h * 128 + lane + 32 * i] =
        (bf16)(v[i] * rs * w[lane + 32 * i] * sg);
  }
}

// ---------------------------------------------------------------------------
// launcher
// ---------------------------------------------------------------------------
extern "C" void kernel_launch(void* const* d_in, const int* in_sizes, int n_in,
                              void* d_out, int out_size, void* d_ws, size_t ws_size,
                              hipStream_t stream) {
  (void)in_sizes; (void)n_in; (void)out_size; (void)ws_size;
  const float* x = (const float*)d_in[0];
  const float* qkv_w = (const float*)d_in[1];
  const float* g_w = (const float*)d_in[2];
  const float* b_w = (const float*)d_in[3];
  const float* gk_w = (const float*)d_in[4];
  const float* conv_w = (const float*)d_in[5];
  const float* onw = (const float*)d_in[6];
  const float* o_w = (const float*)d_in[7];
  float* out = (float*)d_out;
  char* ws = (char*)d_ws;

  // workspace layout (bytes); aliases are valid given launch order below
  bf16* xbf = (bf16*)(ws + 0);            // 16 MiB (later reused as ybuf)
  bf16* wqkv = (bf16*)(ws + 16777216);    // 12 MiB
  bf16* wg = (bf16*)(ws + 29360128);      // 8 MiB
  bf16* wo = (bf16*)(ws + 37748736);      // 8 MiB
  float* qkv = (float*)(ws + 46137344);   // 48 MiB (later q/k/v)
  float* qbuf = (float*)(ws + 46137344);  // 32 MiB
  float* kbuf = (float*)(ws + 79691776);  // 8 MiB
  float* vbuf = (float*)(ws + 88080384);  // 8 MiB
  float* qkvc = (float*)(ws + 96468992);  // 48 MiB (later gbuf)
  float* gbuf = (float*)(ws + 96468992);  // 32 MiB
  float* gabuf = (float*)(ws + 146800640);
  float* bbuf = (float*)(ws + 147062784);
  float* obuf = (float*)(ws + 147324928);  // 32 MiB  (total ~173 MiB)
  bf16* ybuf = (bf16*)(ws + 0);

  cvt_kernel<<<8192, 256, 0, stream>>>(x, xbf, 2097152);
  cvt_kernel<<<6144, 256, 0, stream>>>(qkv_w, wqkv, 1572864);
  cvt_kernel<<<4096, 256, 0, stream>>>(g_w, wg, 1048576);
  cvt_kernel<<<4096, 256, 0, stream>>>(o_w, wo, 1048576);
  gates_kernel<<<512, 256, 0, stream>>>(x, gk_w, b_w, gabuf, bbuf);
  gemm_bf16<<<dim3(24, 32), 256, 0, stream>>>(xbf, wqkv, qkv, 4096, 3072, 2048);
  conv_silu<<<49152, 256, 0, stream>>>(qkv, conv_w, qkvc);
  prep_kernel<<<12288, 256, 0, stream>>>(qkvc, qbuf, kbuf, vbuf);
  gemm_bf16<<<dim3(16, 32), 256, 0, stream>>>(xbf, wg, gbuf, 4096, 2048, 2048);
  scan_kernel<<<32, 256, 0, stream>>>(qbuf, kbuf, vbuf, gabuf, bbuf, obuf);
  rms_gate_kernel<<<8192, 256, 0, stream>>>(obuf, gbuf, onw, ybuf);
  gemm_bf16<<<dim3(16, 32), 256, 0, stream>>>(ybuf, wo, out, 4096, 2048, 2048);
}
